// LPKT_28312424415250
// MI455X (gfx1250) — compile-verified, matrix operand order checked by reading, not run on hardware
//
#include <hip/hip_runtime.h>
#include <hip/hip_bf16.h>
#include <cstddef>

// ---------------------------------------------------------------------------
// LPKT recurrence for MI455X (gfx1250).
// Kernel A: all_learning = concat(e_emb, at_emb, a_rep) @ W1 + b1  (WMMA bf16)
// Kernel B: 99-step scan, one workgroup per batch element, h-state in LDS,
//           h_pre @ W4h via v_wmma_f32_16x16x32_bf16; B operand (W4h^T) held
//           in registers for the whole scan.
// ---------------------------------------------------------------------------

typedef __attribute__((ext_vector_type(16))) __bf16 v16bf;
typedef __attribute__((ext_vector_type(8)))  __bf16 v8bf;
typedef __attribute__((ext_vector_type(8)))  float  v8f;

namespace lpkt {

constexpr int Bc   = 64;
constexpr int Sc   = 100;
constexpr int NQ1  = 129;   // N_Q + 1
constexpr int DK   = 128;

// padded strides (bank-conflict avoidance: 136 halves = 272B -> +4 dwords/lane)
constexpr int HROWS = 144;  // 129 padded to 9 tiles of 16
constexpr int HS32  = 132;  // f32 h row stride (floats)
constexpr int HSB   = 136;  // bf16 h row stride (halves)
constexpr int WTS   = 136;  // bf16 W4h^T row stride (halves)

constexpr int AKS   = 392;  // kernel A: bf16 row stride for K=384 tiles

__device__ __forceinline__ float sigm(float x) {
    return 1.0f / (1.0f + __expf(-x));
}
__device__ __forceinline__ float tanh_(float x) {
    float e = __expf(-2.0f * x);
    return (1.0f - e) / (1.0f + e);
}

// Load one WMMA A/B operand (16x16x32 bf16 layout) from LDS, row-major.
// p points at (row, kbase) where row = tile_row0 + (lane&15),
// kbase = chunk_k0 + ((lane>>4)<<3). Elements 0..7 = K kbase..kbase+7,
// elements 8..15 = K kbase+16..kbase+23  ->  two ds_load_b128.
__device__ __forceinline__ v16bf ld_ab(const __bf16* p) {
    v8bf lo = *(const v8bf*)(p);
    v8bf hi = *(const v8bf*)(p + 16);
    v16bf r;
#pragma unroll
    for (int i = 0; i < 8; ++i) { r[i] = lo[i]; r[i + 8] = hi[i]; }
    return r;
}

// ---------------------------------------------------------------------------
// Kernel A: AL[bs, f] = (concat feats)(bs, :) @ W1 + b1, bf16 WMMA.
// grid = 400 blocks (16 rows each), block = 256 (8 waves, one 16-col tile each)
// ---------------------------------------------------------------------------
__global__ __launch_bounds__(256)
void all_learning_kernel(const int* __restrict__ e_data,
                         const int* __restrict__ a_data,
                         const int* __restrict__ at_data,
                         const float* __restrict__ E_e,
                         const float* __restrict__ E_at,
                         const float* __restrict__ W1,
                         const float* __restrict__ b1,
                         float* __restrict__ AL) {
    extern __shared__ char smem[];
    __bf16* sA = (__bf16*)smem;          // 16  x AKS
    __bf16* sW = sA + 16 * AKS;          // 128 x AKS (W1 transposed: [n][k])

    const int ROWS = Bc * Sc;            // 6400
    const int tid  = threadIdx.x;
    const int lane = tid & 31;
    const int wv   = tid >> 5;
    const int m0   = blockIdx.x * 16;

    // Build gathered A tile (bf16)
    for (int idx = tid; idx < 16 * 384; idx += 256) {
        int r = idx / 384, k = idx - r * 384;
        int bs = m0 + r;
        float v = 0.0f;
        if (bs < ROWS) {
            if (k < 128)      v = E_e [(size_t)e_data [bs] * DK + k];
            else if (k < 256) v = E_at[(size_t)at_data[bs] * DK + (k - 128)];
            else              v = (float)a_data[bs];
        }
        sA[r * AKS + k] = (__bf16)v;
    }
    // Stage W1 transposed: sW[n][k] = W1[k][n]
    for (int idx = tid; idx < 384 * 128; idx += 256) {
        int k = idx >> 7, n = idx & 127;
        sW[n * AKS + k] = (__bf16)W1[idx];
    }
    __syncthreads();

    const int nt    = wv;                 // column tile 0..7
    const int lrow  = lane & 15;
    const int khalf = (lane >> 4) << 3;   // 0 or 8
    v8f acc = {};
#pragma unroll
    for (int kc = 0; kc < 12; ++kc) {     // K = 384 = 12 * 32
        int k0 = kc * 32 + khalf;
        v16bf a = ld_ab(sA + lrow * AKS + k0);
        v16bf b = ld_ab(sW + (nt * 16 + lrow) * AKS + k0);
        acc = __builtin_amdgcn_wmma_f32_16x16x32_bf16(false, a, false, b,
                                                      (short)0, acc, false, false);
    }
    const int fcol = nt * 16 + lrow;
    const float bb = b1[fcol];
    const int mb   = (lane >> 4) << 3;
#pragma unroll
    for (int v = 0; v < 8; ++v) {
        int bs = m0 + mb + v;             // D layout: M = v (+8 for hi lanes)
        if (bs < ROWS) AL[(size_t)bs * DK + fcol] = acc[v] + bb;
    }
}

// ---------------------------------------------------------------------------
// Kernel B: full 99-step recurrence; one workgroup per batch element.
// ---------------------------------------------------------------------------
__global__ __launch_bounds__(256)
void recurrence_kernel(const int* __restrict__ e_data,
                       const int* __restrict__ it_data,
                       const float* __restrict__ q_matrix,
                       const float* __restrict__ h0,
                       const float* __restrict__ E_e,
                       const float* __restrict__ E_it,
                       const float* __restrict__ W2, const float* __restrict__ b2,
                       const float* __restrict__ W3, const float* __restrict__ b3,
                       const float* __restrict__ W4, const float* __restrict__ b4,
                       const float* __restrict__ W5, const float* __restrict__ b5,
                       const float* __restrict__ AL,
                       float* __restrict__ out) {
    extern __shared__ char smem[];
    float*  sH    = (float*)smem;               // HROWS x HS32 (f32 h state)
    float*  sQ    = sH    + HROWS * HS32;       // 2 x 144 (q ping-pong)
    float*  sFeat = sQ    + 2 * 144;            // 512
    float*  sLG   = sFeat + 512;                // 128
    float*  sBase = sLG   + 128;                // 128
    float*  sHt   = sBase + 128;                // 128 (h_tilde)
    float*  sEn   = sHt   + 128;                // 128 (e_next)
    float*  sRed  = sEn   + 128;                // 128 (reduction)
    float*  sP2   = sRed  + 128;                // 256 (split-K partials)
    float*  sP3   = sP2   + 256;                // 256
    __bf16* sWt   = (__bf16*)(sP3 + 256);       // 128 x WTS  (W4h^T bf16)
    __bf16* sHb   = sWt + 128 * WTS;            // 2 x HROWS x HSB (bf16 h mirror)

    const int b    = blockIdx.x;
    const int tid  = threadIdx.x;
    const int lane = tid & 31;
    const int wv   = tid >> 5;

    // ---- preamble -------------------------------------------------------
    // W4h transposed bf16: sWt[f][d] = W4[d*128 + f]   (W4h = W4 rows 0..127)
    for (int idx = tid; idx < 128 * 128; idx += 256) {
        int d = idx >> 7, f = idx & 127;
        sWt[f * WTS + d] = (__bf16)W4[idx];
    }
    // h init (both bf16 buffers; pad rows zeroed and never rewritten)
    for (int idx = tid; idx < HROWS * DK; idx += 256) {
        int r = idx >> 7, c = idx & 127;
        float v = (r < NQ1) ? h0[r * DK + c] : 0.0f;
        sH[r * HS32 + c] = v;
        __bf16 bv = (__bf16)v;
        sHb[r * HSB + c] = bv;
        sHb[HROWS * HSB + r * HSB + c] = bv;
    }
    {   // q row for t = 0 into buffer 0
        int e0 = e_data[b * Sc + 0];
        for (int idx = tid; idx < HROWS; idx += 256)
            sQ[idx] = (idx < NQ1) ? q_matrix[(size_t)e0 * NQ1 + idx] : 0.0f;
    }
    if (tid == 0) out[b * Sc + 0] = 0.0f;
    __syncthreads();

    // ht0 = (q0 . h0) / sum(q0)
    if (tid < DK) {
        float num = 0.0f, den = 0.0f;
        for (int n = 0; n < NQ1; ++n) {
            float q = sQ[n];
            num = fmaf(q, sH[n * HS32 + tid], num);
            den += q;
        }
        sHt[tid] = num / den;
    }
    __syncthreads();

    // WMMA geometry for this wave; B operand (W4h^T) is invariant across the
    // whole scan -> load its 4 K-chunks into registers once.
    const int nt    = wv;                 // column tile 0..7
    const int lrow  = lane & 15;
    const int khalf = (lane >> 4) << 3;   // 0 or 8
    const int fcol  = nt * 16 + lrow;
    const int mhi   = (lane >> 4) << 3;
    v16bf wreg[4];
#pragma unroll
    for (int kc = 0; kc < 4; ++kc)
        wreg[kc] = ld_ab(sWt + fcol * WTS + kc * 32 + khalf);

    // ---- scan -----------------------------------------------------------
    for (int t = 0; t < Sc - 1; ++t) {
        float*  qc  = sQ + (t & 1) * 144;
        float*  qn  = sQ + ((t + 1) & 1) * 144;
        __bf16* hbc = sHb + (t & 1) * (HROWS * HSB);
        __bf16* hbn = sHb + ((t + 1) & 1) * (HROWS * HSB);

        const int itrow = it_data[b * Sc + t];
        const int enext = e_data[b * Sc + t + 1];

        // stage feat = [learning_pre | it_emb | learning | h_tilde_pre], q_next
        if (tid < 128) {
            int j = tid;
            sFeat[j]       = (t == 0) ? 0.0f : AL[((size_t)b * Sc + (t - 1)) * DK + j];
            sFeat[256 + j] = AL[((size_t)b * Sc + t) * DK + j];
        } else {
            int j = tid - 128;
            sFeat[128 + j] = E_it[(size_t)itrow * DK + j];
            sFeat[384 + j] = sHt[j];
        }
        for (int idx = tid; idx < HROWS; idx += 256)
            qn[idx] = (idx < NQ1) ? q_matrix[(size_t)enext * NQ1 + idx] : 0.0f;
        __syncthreads();

        // feat @ W2 / W3, split-K across all 256 threads (each does K/2 = 256)
        {
            int j  = tid & 127;
            int k0 = (tid >> 7) << 8;     // 0 or 256
            float z2 = 0.0f, z3 = 0.0f;
            for (int k = k0; k < k0 + 256; ++k) {
                float fv = sFeat[k];
                z2 = fmaf(fv, W2[k * 128 + j], z2);
                z3 = fmaf(fv, W3[k * 128 + j], z3);
            }
            sP2[tid] = z2;
            sP3[tid] = z3;
        }
        __syncthreads();
        if (tid < 128) {
            int j = tid;
            float z2 = sP2[j] + sP2[j + 128] + b2[j];
            float z3 = sP3[j] + sP3[j + 128] + b3[j];
            sLG[j] = sigm(z3) * (tanh_(z2) + 1.0f) * 0.5f;
        } else {
            int j = tid - 128;
            float acc = b4[j];
            for (int k = 0; k < 128; ++k)
                acc = fmaf(sFeat[128 + k], W4[(256 + k) * 128 + j], acc); // W4i
            sBase[j] = acc;
        }
        __syncthreads();
        if (tid < 128) {
            int j = tid;
            float acc = sBase[j];
            for (int k = 0; k < 128; ++k)
                acc = fmaf(sLG[k], W4[(128 + k) * 128 + j], acc);        // W4l
            sBase[j] = acc;
        }
        __syncthreads();

        // big GEMM h_pre @ W4h (WMMA) fused with gate + h update.
        // wave wv owns column tile nt; mt sweeps 9 row tiles (144 rows).
        {
            const float lgv = sLG[fcol];
            const float bas = sBase[fcol];
#pragma unroll 3
            for (int mt = 0; mt < 9; ++mt) {
                v8f acc = {};
#pragma unroll
                for (int kc = 0; kc < 4; ++kc) {       // K = 128 = 4 * 32
                    v16bf a = ld_ab(hbc + (mt * 16 + lrow) * HSB + kc * 32 + khalf);
                    acc = __builtin_amdgcn_wmma_f32_16x16x32_bf16(
                        false, a, false, wreg[kc], (short)0, acc, false, false);
                }
                const int mbase = mt * 16 + mhi;
#pragma unroll
                for (int v = 0; v < 8; ++v) {
                    int m = mbase + v;
                    if (m < NQ1) {
                        float gf   = sigm(acc[v] + bas);
                        float hold = sH[m * HS32 + fcol];
                        float hnew = fmaf(gf, hold, qc[m] * lgv);
                        sH[m * HS32 + fcol]  = hnew;
                        hbn[m * HSB + fcol]  = (__bf16)hnew;
                    }
                }
            }
        }
        __syncthreads();

        // h_tilde = (q_next . h) / sum(q_next); stage e_next in parallel
        if (tid < 128) {
            float num = 0.0f, den = 0.0f;
            for (int n = 0; n < NQ1; ++n) {
                float q = qn[n];
                num = fmaf(q, sH[n * HS32 + tid], num);
                den += q;
            }
            sHt[tid] = num / den;
        } else {
            int j = tid - 128;
            sEn[j] = E_e[(size_t)enext * DK + j];
        }
        __syncthreads();

        // y = mean(sigmoid([e_next | h_tilde] @ W5 + b5)), split-K then reduce
        {
            int j    = tid & 127;
            int half = tid >> 7;
            float u  = 0.0f;
            if (half == 0) {
                for (int k = 0; k < 128; ++k)
                    u = fmaf(sEn[k], W5[k * 128 + j], u);
            } else {
                for (int k = 0; k < 128; ++k)
                    u = fmaf(sHt[k], W5[(128 + k) * 128 + j], u);
            }
            sP2[tid] = u;
        }
        __syncthreads();
        if (tid < 128)
            sRed[tid] = sigm(sP2[tid] + sP2[tid + 128] + b5[tid]);
        __syncthreads();
        if (tid < 32) {
            float s = sRed[tid] + sRed[tid + 32] + sRed[tid + 64] + sRed[tid + 96];
#pragma unroll
            for (int off = 16; off > 0; off >>= 1)
                s += __shfl_xor(s, off, 32);
            if (tid == 0) out[b * Sc + t + 1] = s * (1.0f / (float)DK);
        }
        __syncthreads();
    }
}

} // namespace lpkt

extern "C" void kernel_launch(void* const* d_in, const int* in_sizes, int n_in,
                              void* d_out, int out_size, void* d_ws, size_t ws_size,
                              hipStream_t stream) {
    using namespace lpkt;
    const int*   e_data   = (const int*)  d_in[0];
    const int*   a_data   = (const int*)  d_in[1];
    const int*   it_data  = (const int*)  d_in[2];
    const int*   at_data  = (const int*)  d_in[3];
    const float* q_matrix = (const float*)d_in[4];
    const float* h0       = (const float*)d_in[5];
    const float* E_e      = (const float*)d_in[6];
    const float* E_at     = (const float*)d_in[7];
    const float* E_it     = (const float*)d_in[8];
    const float* W1 = (const float*)d_in[9];  const float* b1 = (const float*)d_in[10];
    const float* W2 = (const float*)d_in[11]; const float* b2 = (const float*)d_in[12];
    const float* W3 = (const float*)d_in[13]; const float* b3 = (const float*)d_in[14];
    const float* W4 = (const float*)d_in[15]; const float* b4 = (const float*)d_in[16];
    const float* W5 = (const float*)d_in[17]; const float* b5 = (const float*)d_in[18];
    float* out = (float*)d_out;
    float* AL  = (float*)d_ws;   // 64*100*128 f32 = 3.28 MB scratch

    const size_t smemA = (size_t)(16 + 128) * AKS * sizeof(__bf16);          // ~113 KB
    const size_t smemR = (size_t)(HROWS * HS32 + 2 * 144 + 512 + 5 * 128 + 512) * 4
                       + (size_t)(128 * WTS + 2 * HROWS * HSB) * 2;          // ~197 KB

    // Host-side, deterministic, non-stream ops (safe under graph capture).
    hipFuncSetAttribute((const void*)all_learning_kernel,
                        hipFuncAttributeMaxDynamicSharedMemorySize, (int)smemA);
    hipFuncSetAttribute((const void*)recurrence_kernel,
                        hipFuncAttributeMaxDynamicSharedMemorySize, (int)smemR);

    const int rows = Bc * Sc;                       // 6400
    const int gridA = (rows + 15) / 16;             // 400
    all_learning_kernel<<<gridA, 256, smemA, stream>>>(
        e_data, a_data, at_data, E_e, E_at, W1, b1, AL);

    recurrence_kernel<<<Bc, 256, smemR, stream>>>(
        e_data, it_data, q_matrix, h0, E_e, E_it,
        W2, b2, W3, b3, W4, b4, W5, b5, AL, out);
}
